// RNN_74019466379629
// MI455X (gfx1250) — matrix operand order repaired and checked by reading, not compiled, not run on hardware
//
#include <hip/hip_runtime.h>

typedef __attribute__((ext_vector_type(16))) __bf16 v16bf;
typedef __attribute__((ext_vector_type(8)))  float  v8f;

#define BB 128
#define TT 256
#define DD 512
#define UU 1024
#define CC 10

#define WMMA_BF16(A, B, C) \
  __builtin_amdgcn_wmma_f32_16x16x32_bf16(false, (A), false, (B), (short)0, (C), false, false)

static __device__ __forceinline__ __bf16 f2bf(float f) {
  union { float f; unsigned u; } v; v.f = f;
  unsigned r = v.u + 0x7FFFu + ((v.u >> 16) & 1u);   // round to nearest even
  union { unsigned short s; __bf16 b; } o; o.s = (unsigned short)(r >> 16);
  return o.b;
}

static __device__ __forceinline__ float sigmoidf_(float x) {
  return 1.0f / (1.0f + __expf(-x));
}

// Pack x [B,T,D] fp32 -> bf16 A-fragments: [T][B/16][D/32][512]
// A 16x32 bf16 layout: lane(0..15)=row M, lane half selects K subset;
// element j of lane: K = (j/8)*16 + half*8 + (j%8)
__global__ void pack_x_kernel(const float* __restrict__ x, __bf16* __restrict__ dst) {
  int idx = blockIdx.x * blockDim.x + threadIdx.x;
  if (idx >= BB * TT * DD) return;
  int j    = idx & 15;
  int lane = (idx >> 4) & 31;
  int frag = idx >> 9;
  int tk = frag & 15;          // D/32 = 16
  int tmp = frag >> 4;
  int tm = tmp & 7;            // B/16 = 8
  int t  = tmp >> 3;
  int half = lane >> 4;
  int r    = lane & 15;
  int kk = ((j >> 3) << 4) + (half << 3) + (j & 7);
  int b = tm * 16 + r;
  int d = tk * 32 + kk;
  dst[idx] = f2bf(x[((size_t)b * TT + t) * DD + d]);
}

// Pack weight [K,N] fp32 (row-major) -> bf16 B-fragments: [N/16][K/32][512]
// B 32x16 bf16 layout: lane(0..15)=col N, half selects K subset; K = (j/8)*16 + half*8 + (j%8)
__global__ void pack_w_kernel(const float* __restrict__ w, __bf16* __restrict__ dst,
                              int K, int N, int kTiles) {
  int idx = blockIdx.x * blockDim.x + threadIdx.x;
  if (idx >= K * N) return;
  int j    = idx & 15;
  int lane = (idx >> 4) & 31;
  int frag = idx >> 9;
  int tk = frag % kTiles;
  int tn = frag / kTiles;
  int half = lane >> 4;
  int col  = lane & 15;
  int k = tk * 32 + ((j >> 3) << 4) + (half << 3) + (j & 7);
  int n = tn * 16 + col;
  dst[idx] = f2bf(w[(size_t)k * N + n]);
}

// Software-pipelined fragment GEMM: 2 m-tiles x 4 n-tiles per wave, K = KT*32.
// Prefetches k-step ks+1 into fresh registers before consuming ks, so the
// s_wait before the WMMA block only covers the older load group.
template <int KT>
static __device__ __forceinline__ void gemm_tiles(const __bf16* __restrict__ a0p,
                                                  const __bf16* __restrict__ a1p,
                                                  const __bf16* __restrict__ bp,
                                                  v8f (&acc)[2][4]) {
  constexpr size_t NS = (size_t)KT * 512;   // n-tile stride in the packed layout
  v16bf a0 = *(const v16bf*)a0p;
  v16bf a1 = *(const v16bf*)a1p;
  v16bf b0 = *(const v16bf*)(bp);
  v16bf b1 = *(const v16bf*)(bp + NS);
  v16bf b2 = *(const v16bf*)(bp + 2 * NS);
  v16bf b3 = *(const v16bf*)(bp + 3 * NS);
  #pragma unroll 2
  for (int ks = 0; ks < KT - 1; ++ks) {
    const size_t off = (size_t)(ks + 1) * 512;
    v16bf na0 = *(const v16bf*)(a0p + off);
    v16bf na1 = *(const v16bf*)(a1p + off);
    v16bf nb0 = *(const v16bf*)(bp + off);
    v16bf nb1 = *(const v16bf*)(bp + off + NS);
    v16bf nb2 = *(const v16bf*)(bp + off + 2 * NS);
    v16bf nb3 = *(const v16bf*)(bp + off + 3 * NS);
    acc[0][0] = WMMA_BF16(a0, b0, acc[0][0]);
    acc[0][1] = WMMA_BF16(a0, b1, acc[0][1]);
    acc[0][2] = WMMA_BF16(a0, b2, acc[0][2]);
    acc[0][3] = WMMA_BF16(a0, b3, acc[0][3]);
    acc[1][0] = WMMA_BF16(a1, b0, acc[1][0]);
    acc[1][1] = WMMA_BF16(a1, b1, acc[1][1]);
    acc[1][2] = WMMA_BF16(a1, b2, acc[1][2]);
    acc[1][3] = WMMA_BF16(a1, b3, acc[1][3]);
    a0 = na0; a1 = na1; b0 = nb0; b1 = nb1; b2 = nb2; b3 = nb3;
  }
  acc[0][0] = WMMA_BF16(a0, b0, acc[0][0]);
  acc[0][1] = WMMA_BF16(a0, b1, acc[0][1]);
  acc[0][2] = WMMA_BF16(a0, b2, acc[0][2]);
  acc[0][3] = WMMA_BF16(a0, b3, acc[0][3]);
  acc[1][0] = WMMA_BF16(a1, b0, acc[1][0]);
  acc[1][1] = WMMA_BF16(a1, b1, acc[1][1]);
  acc[1][2] = WMMA_BF16(a1, b2, acc[1][2]);
  acc[1][3] = WMMA_BF16(a1, b3, acc[1][3]);
}

// One LSTM timestep. grid = (B/32, U/64), block = 128 (4 waves, wave g = gate g).
// Each wave: 32 rows x 64 u-cols of one gate. Fused K loop: 512 (x_t) + 1024 (h).
__global__ __launch_bounds__(128)
void lstm_step_kernel(int t,
                      const __bf16* __restrict__ xfrag,
                      const __bf16* __restrict__ wkf,
                      const __bf16* __restrict__ wrf,
                      const float*  __restrict__ bias,
                      const __bf16* __restrict__ hcur,
                      __bf16*       __restrict__ hnext,
                      float*        __restrict__ cstate,
                      float*        __restrict__ hf32) {
  __shared__ float lds[4 * 32 * 65];   // [gate][row 0..31][col(+pad)]
  const int tm   = blockIdx.x;         // 0..3   covers rows tm*32 .. tm*32+31
  const int uc   = blockIdx.y;         // 0..15  u-chunk of 64
  const int wave = threadIdx.x >> 5;   // gate id 0..3 (i,f,g,o)
  const int lane = threadIdx.x & 31;
  const int laneoff = lane * 16;

  v8f acc[2][4] = {};                       // [m-tile][n-tile]
  const int tnbase = wave * 64 + uc * 4;    // global n-tile base (N/16 units)

  // ---- input projection: K = 512 (16 k-tiles). m-tile stride in xfrag = 16*512 ----
  {
    const __bf16* a0p = xfrag + (size_t)((t * 8 + tm * 2) * 16) * 512 + laneoff;
    const __bf16* a1p = a0p + (size_t)16 * 512;
    const __bf16* bp  = wkf + (size_t)tnbase * 16 * 512 + laneoff;
    gemm_tiles<16>(a0p, a1p, bp, acc);
  }
  // ---- recurrent: K = 1024 (32 k-tiles). m-tile stride in hfrag = 32*512 ----
  {
    const __bf16* a0p = hcur + (size_t)(tm * 2) * 32 * 512 + laneoff;
    const __bf16* a1p = a0p + (size_t)32 * 512;
    const __bf16* bp  = wrf + (size_t)tnbase * 32 * 512 + laneoff;
    gemm_tiles<32>(a0p, a1p, bp, acc);
  }

  // ---- stage gate pre-activations to LDS (C/D layout: M = r + half*8, N = lane%16) ----
  const int halfc = lane >> 4;
  const int col   = lane & 15;
  #pragma unroll
  for (int mh = 0; mh < 2; ++mh) {
    #pragma unroll
    for (int nt = 0; nt < 4; ++nt) {
      #pragma unroll
      for (int r = 0; r < 8; ++r) {
        lds[(wave * 32 + mh * 16 + halfc * 8 + r) * 65 + nt * 16 + col] = acc[mh][nt][r];
      }
    }
  }
  __syncthreads();

  // ---- pointwise gate math: 32x64 elements, 16 per thread ----
  #pragma unroll
  for (int q = 0; q < 16; ++q) {
    int e = threadIdx.x * 16 + q;
    int m = e >> 6;       // 0..31
    int u = e & 63;
    int gu = uc * 64 + u;
    float zi = lds[(0 * 32 + m) * 65 + u] + bias[0 * UU + gu];
    float zf = lds[(1 * 32 + m) * 65 + u] + bias[1 * UU + gu];
    float zg = lds[(2 * 32 + m) * 65 + u] + bias[2 * UU + gu];
    float zo = lds[(3 * 32 + m) * 65 + u] + bias[3 * UU + gu];
    float ig = sigmoidf_(zi);
    float fg = sigmoidf_(zf);
    float gg = tanhf(zg);
    float og = sigmoidf_(zo);
    int gm = tm * 32 + m;
    float cold = cstate[gm * UU + gu];
    float cn = fg * cold + ig * gg;
    cstate[gm * UU + gu] = cn;
    float hn = og * tanhf(cn);
    hf32[gm * UU + gu] = hn;
    // write h in bf16 A-fragment layout for next step's recurrent GEMM
    int tk    = gu >> 5;
    int kk    = gu & 31;
    int group = kk >> 4;
    int rem   = kk & 15;
    int halfb = rem >> 3;
    int jj    = group * 8 + (rem & 7);
    int lane2 = halfb * 16 + (m & 15);
    int tmf   = gm >> 4;                 // global m-tile index
    hnext[(size_t)(tmf * 32 + tk) * 512 + lane2 * 16 + jj] = f2bf(hn);
  }
}

// Tiny classifier + softmax: logits = h_last @ w_cls + b_cls. One block per batch row.
__global__ __launch_bounds__(64)
void cls_kernel(const float* __restrict__ h, const float* __restrict__ w,
                const float* __restrict__ b, float* __restrict__ out) {
  __shared__ float part[64 * CC];
  __shared__ float lg[CC];
  int bidx = blockIdx.x;
  int tid  = threadIdx.x;
  float acc[CC];
  #pragma unroll
  for (int c = 0; c < CC; ++c) acc[c] = 0.0f;
  for (int u = tid; u < UU; u += 64) {
    float hv = h[bidx * UU + u];
    #pragma unroll
    for (int c = 0; c < CC; ++c) acc[c] += hv * w[u * CC + c];
  }
  #pragma unroll
  for (int c = 0; c < CC; ++c) part[tid * CC + c] = acc[c];
  __syncthreads();
  if (tid < CC) {
    float s = 0.0f;
    for (int j = 0; j < 64; ++j) s += part[j * CC + tid];
    lg[tid] = s + b[tid];
  }
  __syncthreads();
  if (tid == 0) {
    float mx = lg[0];
    for (int c = 1; c < CC; ++c) mx = fmaxf(mx, lg[c]);
    float e[CC];
    float sum = 0.0f;
    for (int c = 0; c < CC; ++c) { e[c] = __expf(lg[c] - mx); sum += e[c]; }
    float inv = 1.0f / sum;
    for (int c = 0; c < CC; ++c) out[bidx * CC + c] = e[c] * inv;
  }
}

extern "C" void kernel_launch(void* const* d_in, const int* in_sizes, int n_in,
                              void* d_out, int out_size, void* d_ws, size_t ws_size,
                              hipStream_t stream) {
  (void)in_sizes; (void)n_in; (void)out_size; (void)ws_size;
  const float* x       = (const float*)d_in[0];
  const float* kernelw = (const float*)d_in[1];
  const float* rkernel = (const float*)d_in[2];
  const float* bias    = (const float*)d_in[3];
  const float* wcls    = (const float*)d_in[4];
  const float* bcls    = (const float*)d_in[5];
  float* out = (float*)d_out;

  // workspace layout (~45.5 MiB total)
  char* ws = (char*)d_ws;
  size_t o = 0;
  __bf16* xfrag = (__bf16*)(ws + o); o += (size_t)BB * TT * DD * 2;   // 32 MiB
  __bf16* wkf   = (__bf16*)(ws + o); o += (size_t)DD * 4 * UU * 2;    // 4 MiB
  __bf16* wrf   = (__bf16*)(ws + o); o += (size_t)UU * 4 * UU * 2;    // 8 MiB
  __bf16* hA    = (__bf16*)(ws + o); o += (size_t)BB * UU * 2;        // 256 KiB
  __bf16* hB    = (__bf16*)(ws + o); o += (size_t)BB * UU * 2;        // 256 KiB
  float*  cst   = (float*)(ws + o);  o += (size_t)BB * UU * 4;        // 512 KiB
  float*  hf    = (float*)(ws + o);  o += (size_t)BB * UU * 4;        // 512 KiB

  hipMemsetAsync(hA,  0, (size_t)BB * UU * 2, stream);   // h0 = 0 (bf16 zero = 0x0000)
  hipMemsetAsync(cst, 0, (size_t)BB * UU * 4, stream);   // c0 = 0

  {
    int total = BB * TT * DD;
    pack_x_kernel<<<(total + 255) / 256, 256, 0, stream>>>(x, xfrag);
  }
  {
    int total = DD * 4 * UU;
    pack_w_kernel<<<(total + 255) / 256, 256, 0, stream>>>(kernelw, wkf, DD, 4 * UU, DD / 32);
  }
  {
    int total = UU * 4 * UU;
    pack_w_kernel<<<(total + 255) / 256, 256, 0, stream>>>(rkernel, wrf, UU, 4 * UU, UU / 32);
  }

  __bf16* hcur = hA;
  __bf16* hnext = hB;
  dim3 grid(BB / 32, UU / 64);   // 4 x 16 = 64 workgroups per step
  for (int t = 0; t < TT; ++t) {
    lstm_step_kernel<<<grid, 128, 0, stream>>>(t, xfrag, wkf, wrf, bias,
                                               hcur, hnext, cst, hf);
    __bf16* tmp = hcur; hcur = hnext; hnext = tmp;
  }

  cls_kernel<<<BB, 64, 0, stream>>>(hf, wcls, bcls, out);
}
